// LSTMModel_57818849739401
// MI455X (gfx1250) — matrix operand order repaired and checked by reading, not compiled, not run on hardware
//
#include <hip/hip_runtime.h>
#include <hip/hip_bf16.h>

typedef __attribute__((ext_vector_type(16))) __bf16 v16bf;
typedef __attribute__((ext_vector_type(8)))  float  v8f;

#define HDIM 64
#define FDIM 32
#define BATCH 4096
#define TLEN 256
#define GDIM 256   // 4*H

// ---------------------------------------------------------------------------
// LDS layout (dynamic shared, 120 KB per workgroup, 2 waves / 32 rows per WG):
//   [  0K,  16K)  W_ih0 bf16 B-fragments (16 ntiles x 1 ktile)
//   [ 16K,  48K)  W_hh0 bf16 B-fragments (16 ntiles x 2 ktiles)
//   [ 48K,  80K)  W_ih1 bf16 B-fragments (16 ntiles x 2 ktiles)
//   [ 80K, 112K)  W_hh1 bf16 B-fragments (16 ntiles x 2 ktiles)
//   [112K, 120K)  per-wave h staging: wave w -> h0[16][64], h1[16][64] bf16
// B-fragment blob index: (((ntile*nk + ktile)*32 + lane)*16 + e) bf16 elems,
// lane<16: K = kt*32+{0..7,16..23}[e] ; lane>=16: +8 ; col = ntile*16+(lane&15)
// (ISA 7.12.2 B 32x16 16-bit layout)
// ---------------------------------------------------------------------------

#define LOG2E 1.4426950408889634f

__device__ __forceinline__ float fast_sig(float v) {
    // sigmoid(x) = 1 / (1 + 2^(-x*log2(e)))  -> v_mul, v_exp, v_add, v_rcp
    return __builtin_amdgcn_rcpf(1.0f + __builtin_amdgcn_exp2f(-LOG2E * v));
}

__device__ __forceinline__ float fast_tanh(float v) {
#if __has_builtin(__builtin_amdgcn_tanhf)
    return __builtin_amdgcn_tanhf(v);       // native v_tanh_f32 on gfx1250
#else
    // tanh(x) = 1 - 2/(2^(2x*log2e) + 1)
    return 1.0f - 2.0f * __builtin_amdgcn_rcpf(
               1.0f + __builtin_amdgcn_exp2f(2.0f * LOG2E * v));
#endif
}

__device__ __forceinline__ void pack_weight_bfrag(const float* __restrict__ W,
                                                  __bf16* dst, int Kdim,
                                                  int tid, int nthr) {
    const int nk = Kdim >> 5;                 // K-tiles of 32
    const int total = 16 * nk * 32 * 16;      // bf16 elements
    for (int i = tid; i < total; i += nthr) {
        int e    = i & 15;
        int lane = (i >> 4) & 31;
        int rest = i >> 9;
        int kt   = rest % nk;
        int nt   = rest / nk;
        int k    = kt * 32 + ((lane & 16) >> 1) + (e & 7) + ((e & 8) << 1);  // upper lanes: +8
        int n    = nt * 16 + (lane & 15);
        dst[i] = (__bf16)W[n * Kdim + k];
    }
}

// Load one B-fragment (v16bf) from the packed LDS blob.
__device__ __forceinline__ v16bf ldb(const __bf16* base, int nt, int kt,
                                     int nk, int lane) {
    return *(const v16bf*)(base + (size_t)(((nt * nk + kt) * 32 + lane)) * 16);
}

// Load a 16x32 A-fragment from row-major bf16 staging [16][64].
// lanes 0-15: K = kt*32 + {0..7, 16..23}; lanes 16-31: +8.
__device__ __forceinline__ v16bf ldah(const __bf16* stg, int kt, int lane) {
    int row = lane & 15;
    int c0  = kt * 32 + ((lane & 16) ? 8 : 0);
    const __bf16* p = stg + row * 64 + c0;
    v16bf a;
#pragma unroll
    for (int e = 0; e < 8; ++e) {
        a[e]     = p[e];
        a[8 + e] = p[16 + e];
    }
    return a;
}

// Load a 16x32 A-fragment of x (fp32 global), converting to bf16.
__device__ __forceinline__ v16bf ldax(const float* __restrict__ xrow, int lane) {
    int c0 = (lane & 16) ? 8 : 0;
    v16bf a;
#pragma unroll
    for (int e = 0; e < 8; ++e) {
        a[e]     = (__bf16)xrow[c0 + e];
        a[8 + e] = (__bf16)xrow[c0 + 16 + e];
    }
    return a;
}

__device__ __forceinline__ v8f wmma_bf16(v16bf a, v16bf b, v8f c) {
    return __builtin_amdgcn_wmma_f32_16x16x32_bf16(
        /*neg_a=*/false, a, /*neg_b=*/false, b,
        /*c_mod=*/(short)0, c, /*reuse_a=*/false, /*reuse_b=*/false);
}

__global__ void __launch_bounds__(64)
lstm2_fused_kernel(const float* __restrict__ x,
                   const float* __restrict__ Wih0,
                   const float* __restrict__ Whh0,
                   const float* __restrict__ bih0,
                   const float* __restrict__ bhh0,
                   const float* __restrict__ Wih1,
                   const float* __restrict__ Whh1,
                   const float* __restrict__ bih1,
                   const float* __restrict__ bhh1,
                   const float* __restrict__ Wout,
                   const float* __restrict__ bout,
                   float* __restrict__ out) {
    extern __shared__ char smem[];
    __bf16* wih0 = (__bf16*)(smem);
    __bf16* whh0 = (__bf16*)(smem + 16 * 1024);
    __bf16* wih1 = (__bf16*)(smem + 48 * 1024);
    __bf16* whh1 = (__bf16*)(smem + 80 * 1024);

    const int tid  = threadIdx.x;
    const int lane = tid & 31;
    const int wv   = tid >> 5;

    __bf16* h0s = (__bf16*)(smem + 112 * 1024 + wv * 4096);
    __bf16* h1s = h0s + 1024;   // 16*64 bf16 each

    // ---- one-time: convert fp32 weights -> bf16 B-fragments in LDS ----
    pack_weight_bfrag(Wih0, wih0, FDIM, tid, blockDim.x);
    pack_weight_bfrag(Whh0, whh0, HDIM, tid, blockDim.x);
    pack_weight_bfrag(Wih1, wih1, HDIM, tid, blockDim.x);
    pack_weight_bfrag(Whh1, whh1, HDIM, tid, blockDim.x);
    // zero this wave's h staging (h0 and h1 are contiguous: 2048 elems)
    for (int i = lane; i < 2048; i += 32) h0s[i] = (__bf16)0.0f;
    __syncthreads();

    // ---- per-thread fused bias (b_ih + b_hh); applied in elementwise ----
    const int col = lane & 15;
    float bias0[16], bias1[16];
#pragma unroll
    for (int nt = 0; nt < 16; ++nt) {
        bias0[nt] = bih0[nt * 16 + col] + bhh0[nt * 16 + col];
        bias1[nt] = bih1[nt * 16 + col] + bhh1[nt * 16 + col];
    }

    // wave handles rows [rb, rb+16)
    const int rb  = blockIdx.x * 32 + wv * 16;
    const int row = rb + (lane & 15);
    const float* xp = x + (size_t)row * TLEN * FDIM;

    // cell state fp32, 4 column-groups of v8f per layer
    const v8f vzero = (v8f){0.f, 0.f, 0.f, 0.f, 0.f, 0.f, 0.f, 0.f};
    v8f c0[4], c1[4];
#pragma unroll
    for (int j = 0; j < 4; ++j) { c0[j] = vzero; c1[j] = vzero; }

    const int r0 = (lane & 16) ? 8 : 0;     // C/D row base for this lane half

    for (int t = 0; t < TLEN; ++t) {
        const float* xr = xp + t * FDIM;
        if (t + 8 < TLEN) __builtin_prefetch(xp + (t + 8) * FDIM, 0, 0);

        // A-fragments for this step (previous h values)
        v16bf xa   = ldax(xr, lane);
        v16bf h0a0 = ldah(h0s, 0, lane);
        v16bf h0a1 = ldah(h0s, 1, lane);

        // ---------------- layer 0 ----------------
#pragma unroll
        for (int j = 0; j < 4; ++j) {
            v8f acc[4];
#pragma unroll
            for (int q = 0; q < 4; ++q) {
                int nt = q * 4 + j;
                v8f a = vzero;   // bias folded into elementwise below
                a = wmma_bf16(xa,   ldb(wih0, nt, 0, 1, lane), a);
                a = wmma_bf16(h0a0, ldb(whh0, nt, 0, 2, lane), a);
                a = wmma_bf16(h0a1, ldb(whh0, nt, 1, 2, lane), a);
                acc[q] = a;
            }
            float bi = bias0[j],     bf = bias0[4 + j];
            float bg = bias0[8 + j], bo = bias0[12 + j];
            v8f cc = c0[j];
            v8f hh = cc;
#pragma unroll
            for (int e = 0; e < 8; ++e) {
                float ig = fast_sig(acc[0][e] + bi);
                float fg = fast_sig(acc[1][e] + bf);
                float gg = fast_tanh(acc[2][e] + bg);
                float og = fast_sig(acc[3][e] + bo);
                float cv = fg * cc[e] + ig * gg;
                cc[e] = cv;
                hh[e] = og * fast_tanh(cv);
            }
            c0[j] = cc;
            int cidx = j * 16 + (lane & 15);
#pragma unroll
            for (int e = 0; e < 8; ++e)
                h0s[(r0 + e) * 64 + cidx] = (__bf16)hh[e];
        }

        // ---------------- layer 1 (input = fresh h0) ----------------
        v16bf h0n0 = ldah(h0s, 0, lane);
        v16bf h0n1 = ldah(h0s, 1, lane);
        v16bf h1a0 = ldah(h1s, 0, lane);
        v16bf h1a1 = ldah(h1s, 1, lane);
#pragma unroll
        for (int j = 0; j < 4; ++j) {
            v8f acc[4];
#pragma unroll
            for (int q = 0; q < 4; ++q) {
                int nt = q * 4 + j;
                v8f a = vzero;
                a = wmma_bf16(h0n0, ldb(wih1, nt, 0, 2, lane), a);
                a = wmma_bf16(h0n1, ldb(wih1, nt, 1, 2, lane), a);
                a = wmma_bf16(h1a0, ldb(whh1, nt, 0, 2, lane), a);
                a = wmma_bf16(h1a1, ldb(whh1, nt, 1, 2, lane), a);
                acc[q] = a;
            }
            float bi = bias1[j],     bf = bias1[4 + j];
            float bg = bias1[8 + j], bo = bias1[12 + j];
            v8f cc = c1[j];
            v8f hh = cc;
#pragma unroll
            for (int e = 0; e < 8; ++e) {
                float ig = fast_sig(acc[0][e] + bi);
                float fg = fast_sig(acc[1][e] + bf);
                float gg = fast_tanh(acc[2][e] + bg);
                float og = fast_sig(acc[3][e] + bo);
                float cv = fg * cc[e] + ig * gg;
                cc[e] = cv;
                hh[e] = og * fast_tanh(cv);
            }
            c1[j] = cc;
            int cidx = j * 16 + (lane & 15);
#pragma unroll
            for (int e = 0; e < 8; ++e)
                h1s[(r0 + e) * 64 + cidx] = (__bf16)hh[e];
        }
    }

    // ---- output head: out[row] = h1_last . W_out + b_out ----
    if (lane < 16) {
        float s = 0.0f;
#pragma unroll
        for (int n = 0; n < 64; ++n)
            s += (float)h1s[lane * 64 + n] * Wout[n];
        out[rb + lane] = s + bout[0];
    }
}

extern "C" void kernel_launch(void* const* d_in, const int* in_sizes, int n_in,
                              void* d_out, int out_size, void* d_ws, size_t ws_size,
                              hipStream_t stream) {
    const float* x    = (const float*)d_in[0];
    const float* Wih0 = (const float*)d_in[1];
    const float* Whh0 = (const float*)d_in[2];
    const float* bih0 = (const float*)d_in[3];
    const float* bhh0 = (const float*)d_in[4];
    const float* Wih1 = (const float*)d_in[5];
    const float* Whh1 = (const float*)d_in[6];
    const float* bih1 = (const float*)d_in[7];
    const float* bhh1 = (const float*)d_in[8];
    const float* Wout = (const float*)d_in[9];
    const float* bout = (const float*)d_in[10];
    float* out = (float*)d_out;

    const int rows_per_block = 32;               // 2 waves x 16 rows
    const int grid = BATCH / rows_per_block;     // 128 workgroups
    const size_t shmem = 120 * 1024;             // weights 112K + staging 8K

    lstm2_fused_kernel<<<grid, 64, shmem, stream>>>(
        x, Wih0, Whh0, bih0, bhh0, Wih1, Whh1, bih1, bhh1, Wout, bout, out);
}
